// BiLSTMCRF_40621800686086
// MI455X (gfx1250) — matrix operand order repaired and checked by reading, not compiled, not run on hardware
//
#include <hip/hip_runtime.h>
#include <hip/hip_bf16.h>

typedef __attribute__((ext_vector_type(16))) _Float16 v16h;
typedef __attribute__((ext_vector_type(8)))  _Float16 v8h;
typedef __attribute__((ext_vector_type(8)))  float    v8f;

#define B_   64
#define L_   256
#define DM_  256
#define H_   4
#define DK_  64
#define HID2_ 128
#define NT_  20
#define START_ 18
#define STOP_  19
#define NPAD_ 32   // padded tag count for WMMA logits GEMM

// ---------------------------------------------------------------------------
// WMMA fragment loaders (contiguous k order, same for A and B, wave32)
// A: 16x32 f16 tile, row-major source. Element e: M=lane&15,
//    K = (lane>>4)*16 + e  -> two 16B vector loads.
// ---------------------------------------------------------------------------
__device__ inline v16h frag_a_rowmajor(const _Float16* A, int lda, int lane) {
  int lr = lane & 15, hi = lane >> 4;
  const _Float16* p = A + (size_t)lr * lda + hi * 16;
  v8h lo = *(const v8h*)p;
  v8h h2 = *(const v8h*)(p + 8);
  v16h r;
#pragma unroll
  for (int i = 0; i < 8; ++i) { r[i] = lo[i]; r[i + 8] = h2[i]; }
  return r;
}

// B = W^T for a 32x16 tile, where W is (N x K) row-major and points at (n0,k0).
// Element e: N=lane&15, K=(lane>>4)*16+e -> 16 contiguous halfs per lane.
__device__ inline v16h frag_b_transposed(const _Float16* W, int ldw, int lane) {
  int lr = lane & 15, hi = lane >> 4;
  const _Float16* p = W + (size_t)lr * ldw + hi * 16;
  v8h lo = *(const v8h*)p;
  v8h h2 = *(const v8h*)(p + 8);
  v16h r;
#pragma unroll
  for (int i = 0; i < 8; ++i) { r[i] = lo[i]; r[i + 8] = h2[i]; }
  return r;
}

// B row-major (K x N), pointer at (k0,n0). Element e: K=(lane>>4)*16+e, N=lane&15.
__device__ inline v16h frag_b_rowmajor(const _Float16* Bm, int ldb, int lane) {
  int lr = lane & 15, hi = lane >> 4;
  v16h r;
#pragma unroll
  for (int e = 0; e < 16; ++e)
    r[e] = Bm[(size_t)(hi * 16 + e) * ldb + lr];
  return r;
}

__device__ inline float sigmoidf_(float x) { return 1.f / (1.f + __expf(-x)); }

// ---------------------------------------------------------------------------
// Generic C = A(MxK,f16) * W^T(NxK,f16) + bias ; one wave per 16x16 tile.
// ---------------------------------------------------------------------------
__global__ void gemm_xw_kernel(const _Float16* __restrict__ A,
                               const _Float16* __restrict__ W,
                               const float* __restrict__ bias,
                               float* __restrict__ Cf, _Float16* __restrict__ Ch,
                               int M, int N, int K) {
  int lane = threadIdx.x & 31;
  int wave = threadIdx.x >> 5;
  int tile = blockIdx.x * 8 + wave;
  int ntN = N >> 4;
  int ntiles = (M >> 4) * ntN;
  if (tile >= ntiles) return;           // per-wave uniform: EXEC stays all-ones
  int tm = tile / ntN, tn = tile % ntN;
  v8f acc = {};
  for (int k0 = 0; k0 < K; k0 += 32) {
    v16h a = frag_a_rowmajor(A + (size_t)(tm * 16) * K + k0, K, lane);
    v16h b = frag_b_transposed(W + (size_t)(tn * 16) * K + k0, K, lane);
    acc = __builtin_amdgcn_wmma_f32_16x16x32_f16(false, a, false, b,
                                                 (short)0, acc, false, false);
  }
  int lr = lane & 15, hi = lane >> 4;
  int col = tn * 16 + lr;
  float bv = bias ? bias[col] : 0.f;
#pragma unroll
  for (int r = 0; r < 8; ++r) {
    int row = tm * 16 + hi * 8 + r;
    float v = acc[r] + bv;
    if (Cf) Cf[(size_t)row * N + col] = v;
    if (Ch) Ch[(size_t)row * N + col] = (_Float16)v;
  }
}

// ---------------------------------------------------------------------------
// Attention: S[b,h] = Q[b,h] * K[b,h]^T / 8   (per (b,h): 256x256, K=64)
// ---------------------------------------------------------------------------
__global__ void attn_scores_kernel(const _Float16* __restrict__ Q,
                                   const _Float16* __restrict__ Km,
                                   float* __restrict__ S) {
  int lane = threadIdx.x & 31, wave = threadIdx.x >> 5;
  int tile = blockIdx.x * 8 + wave;          // 0..255
  int tm = tile >> 4, tn = tile & 15;
  int bh = blockIdx.y;
  int b = bh >> 2, h = bh & 3;
  const _Float16* Qb = Q + ((size_t)b * L_) * DM_ + h * DK_;
  const _Float16* Kb = Km + ((size_t)b * L_) * DM_ + h * DK_;
  v8f acc = {};
  for (int k0 = 0; k0 < DK_; k0 += 32) {
    v16h a = frag_a_rowmajor(Qb + (size_t)(tm * 16) * DM_ + k0, DM_, lane);
    v16h bfr = frag_b_transposed(Kb + (size_t)(tn * 16) * DM_ + k0, DM_, lane);
    acc = __builtin_amdgcn_wmma_f32_16x16x32_f16(false, a, false, bfr,
                                                 (short)0, acc, false, false);
  }
  int lr = lane & 15, hi = lane >> 4;
  float* Sb = S + (size_t)bh * L_ * L_;
#pragma unroll
  for (int r = 0; r < 8; ++r)
    Sb[(size_t)(tm * 16 + hi * 8 + r) * L_ + tn * 16 + lr] = acc[r] * 0.125f;
}

// Row softmax over 256 cols; one block (8 waves) per row; f16 probs out.
__global__ void softmax_rows_kernel(const float* __restrict__ S,
                                    _Float16* __restrict__ P) {
  size_t row = blockIdx.x;
  int tid = threadIdx.x;
  __shared__ float sm[8], ss[8];
  float v = S[row * 256 + tid];
  float m = v;
#pragma unroll
  for (int o = 16; o > 0; o >>= 1) m = fmaxf(m, __shfl_xor(m, o));
  if ((tid & 31) == 0) sm[tid >> 5] = m;
  __syncthreads();
  float mAll = sm[0];
#pragma unroll
  for (int i = 1; i < 8; ++i) mAll = fmaxf(mAll, sm[i]);
  float e = __expf(v - mAll);
  float s = e;
#pragma unroll
  for (int o = 16; o > 0; o >>= 1) s += __shfl_xor(s, o);
  if ((tid & 31) == 0) ss[tid >> 5] = s;
  __syncthreads();
  float sAll = 0.f;
#pragma unroll
  for (int i = 0; i < 8; ++i) sAll += ss[i];
  P[row * 256 + tid] = (_Float16)(e / sAll);
}

// X[b,h] = P[b,h] * V[b,h]   (per (b,h): 256x64, K=256), B non-transposed.
__global__ void attn_pv_kernel(const _Float16* __restrict__ P,
                               const _Float16* __restrict__ V,
                               _Float16* __restrict__ X) {
  int lane = threadIdx.x & 31, wave = threadIdx.x >> 5;
  int tile = blockIdx.x * 8 + wave;          // 0..63
  int tm = tile >> 2, tn = tile & 3;
  int bh = blockIdx.y;
  int b = bh >> 2, h = bh & 3;
  const _Float16* Pb = P + (size_t)bh * L_ * L_;
  const _Float16* Vb = V + ((size_t)b * L_) * DM_ + h * DK_;
  v8f acc = {};
  for (int k0 = 0; k0 < L_; k0 += 32) {
    v16h a = frag_a_rowmajor(Pb + (size_t)(tm * 16) * L_ + k0, L_, lane);
    v16h bfr = frag_b_rowmajor(Vb + (size_t)k0 * DM_ + tn * 16, DM_, lane);
    acc = __builtin_amdgcn_wmma_f32_16x16x32_f16(false, a, false, bfr,
                                                 (short)0, acc, false, false);
  }
  int lr = lane & 15, hi = lane >> 4;
  _Float16* Xb = X + ((size_t)b * L_) * DM_ + h * DK_;
#pragma unroll
  for (int r = 0; r < 8; ++r)
    Xb[(size_t)(tm * 16 + hi * 8 + r) * DM_ + tn * 16 + lr] = (_Float16)acc[r];
}

// ---------------------------------------------------------------------------
// Persistent BiLSTM recurrence: one block per direction, 512 thr = 16 waves.
// Whh (128 KB f16) is staged into dynamic LDS ONCE via CDNA5 async
// global->LDS copies (ASYNCcnt), then every recurrent step's WMMA reads both
// A (h state) and B (Whh) fragments from LDS. h f16 + c f32 stay resident.
// Total LDS: 48 KB static + 128 KB dynamic = 176 KB (<320 KB/WGP).
// ---------------------------------------------------------------------------
__global__ void lstm_recurrent_kernel(const float* __restrict__ gi_f,
                                      const float* __restrict__ gi_b,
                                      const _Float16* __restrict__ Whh_f16,
                                      const _Float16* __restrict__ Whh_b16,
                                      const float* __restrict__ h0,
                                      const float* __restrict__ c0,
                                      float* __restrict__ grec_ws,
                                      _Float16* __restrict__ lstm_out) {
  int dir = blockIdx.x;
  const float* gi = dir ? gi_b : gi_f;
  const _Float16* Whh = dir ? Whh_b16 : Whh_f16;
  float* g = grec_ws + (size_t)dir * B_ * 512;

  __shared__ __align__(16) _Float16 hsh[B_ * HID2_];
  __shared__ float csh[B_ * HID2_];
  extern __shared__ __align__(16) char dynlds[];
  _Float16* whh_lds = (_Float16*)dynlds;          // 512*128 f16 = 128 KB

  int tid = threadIdx.x;

  // ---- async stage Whh into LDS (8192 x 16B transfers, ASYNCcnt) ----
  {
    unsigned lds_base = (unsigned)(uintptr_t)whh_lds;  // low 32 bits = LDS offset
    const char* gbase = (const char*)Whh;
    for (int chunk = tid; chunk < (512 * HID2_ * 2) / 16; chunk += 512) {
      unsigned laddr = lds_base + (unsigned)chunk * 16u;
      const char* gaddr = gbase + (size_t)chunk * 16;
      asm volatile("global_load_async_to_lds_b128 %0, %1, off"
                   :: "v"(laddr), "v"(gaddr)
                   : "memory");
    }
  }

  // ---- init h, c in LDS ----
  for (int i = tid; i < B_ * HID2_; i += 512) {
    hsh[i] = (_Float16)h0[(size_t)dir * B_ * HID2_ + i];
    csh[i] = c0[(size_t)dir * B_ * HID2_ + i];
  }
  asm volatile("s_wait_asynccnt 0" ::: "memory");
  __syncthreads();

  int lane = tid & 31, wave = tid >> 5;
  int lr = lane & 15, hi = lane >> 4;

  for (int s = 0; s < L_; ++s) {
    int t = dir ? (L_ - 1 - s) : s;
    // ---- recurrent GEMM: (64 x 512) = h(64x128,f16 LDS) * Whh^T(LDS) ----
    for (int tile = wave; tile < 128; tile += 16) {
      int tm = tile >> 5, tn = tile & 31;
      v8f acc = {};
#pragma unroll
      for (int k0 = 0; k0 < HID2_; k0 += 32) {
        v16h a = frag_a_rowmajor(&hsh[(size_t)(tm * 16) * HID2_ + k0], HID2_, lane);
        v16h b = frag_b_transposed(&whh_lds[(size_t)(tn * 16) * HID2_ + k0], HID2_, lane);
        acc = __builtin_amdgcn_wmma_f32_16x16x32_f16(false, a, false, b,
                                                     (short)0, acc, false, false);
      }
#pragma unroll
      for (int r = 0; r < 8; ++r)
        g[(size_t)(tm * 16 + hi * 8 + r) * 512 + tn * 16 + lr] = acc[r];
    }
    __threadfence_block();
    __syncthreads();
    // ---- gate update ----
    for (int i = tid; i < B_ * HID2_; i += 512) {
      int bb = i >> 7, j = i & 127;
      size_t girow = ((size_t)bb * L_ + t) * 512;
      size_t grow = (size_t)bb * 512;
      float gi_i = gi[girow + j]           + g[grow + j];
      float gi_fv = gi[girow + 128 + j]    + g[grow + 128 + j];
      float gi_g = gi[girow + 256 + j]     + g[grow + 256 + j];
      float gi_o = gi[girow + 384 + j]     + g[grow + 384 + j];
      float c = sigmoidf_(gi_fv) * csh[i] + sigmoidf_(gi_i) * tanhf(gi_g);
      float h = sigmoidf_(gi_o) * tanhf(c);
      csh[i] = c;
      hsh[i] = (_Float16)h;
      lstm_out[((size_t)bb * L_ + t) * DM_ + dir * HID2_ + j] = (_Float16)h;
    }
    __threadfence_block();
    __syncthreads();
  }
}

// ---------------------------------------------------------------------------
// CRF: one wave per batch row; lane = tag. logits padded to width 32.
// ---------------------------------------------------------------------------
__global__ void crf_kernel(const float* __restrict__ logitsPad,
                           const int* __restrict__ tags,
                           const int* __restrict__ lengths,
                           const float* __restrict__ trans,
                           float* __restrict__ out) {
  int b = blockIdx.x;
  int j = threadIdx.x;
  __shared__ float prev[NT_];
  if (j < NT_) prev[j] = 0.f;
  int len = lengths[b];
  float real = 0.f;
  __syncthreads();
  for (int t = 0; t < L_; ++t) {
    bool m = (t < len);
    float nv = 0.f;
    if (j < NT_) {
      float lg = logitsPad[((size_t)b * L_ + t) * NPAD_ + j];
      float mx = -1e30f;
#pragma unroll
      for (int i = 0; i < NT_; ++i) mx = fmaxf(mx, prev[i] + trans[i * NT_ + j]);
      float s = 0.f;
#pragma unroll
      for (int i = 0; i < NT_; ++i) s += __expf(prev[i] + trans[i * NT_ + j] - mx);
      nv = mx + __logf(s) + lg;
    }
    __syncthreads();
    if (j < NT_ && m) prev[j] = nv;
    if (j == 0 && m) {
      int cur = tags[b * L_ + t];
      int plab = (t == 0) ? START_ : tags[b * L_ + t - 1];
      real += logitsPad[((size_t)b * L_ + t) * NPAD_ + cur] + trans[plab * NT_ + cur];
    }
    __syncthreads();
  }
  if (j == 0) {
    int last = tags[b * L_ + (len - 1)];
    real += trans[last * NT_ + STOP_];
    float mx = -1e30f;
#pragma unroll
    for (int i = 0; i < NT_; ++i) mx = fmaxf(mx, prev[i] + trans[i * NT_ + STOP_]);
    float s = 0.f;
#pragma unroll
    for (int i = 0; i < NT_; ++i) s += __expf(prev[i] + trans[i * NT_ + STOP_] - mx);
    atomicAdd(out, (mx + __logf(s)) - real);
  }
}

// ---------------------------------------------------------------------------
// Small utility kernels
// ---------------------------------------------------------------------------
__global__ void f2h_kernel(const float* __restrict__ s, _Float16* __restrict__ d, int n) {
  int i = blockIdx.x * 256 + threadIdx.x;
  if (i < n) d[i] = (_Float16)s[i];
}

__global__ void bias_sum_kernel(const float* __restrict__ a, const float* __restrict__ b,
                                float* __restrict__ o, int n) {
  int i = blockIdx.x * 256 + threadIdx.x;
  if (i < n) o[i] = a[i] + b[i];
}

__global__ void pad_wout_kernel(const float* __restrict__ Wout, const float* __restrict__ bout,
                                _Float16* __restrict__ Wp, float* __restrict__ bp) {
  int i = blockIdx.x * 256 + threadIdx.x;   // covers 32*256
  if (i < NPAD_ * DM_) {
    int r = i >> 8;
    Wp[i] = (r < NT_) ? (_Float16)Wout[(size_t)r * DM_ + (i & 255)] : (_Float16)0.f;
  }
  if (i < NPAD_) bp[i] = (i < NT_) ? bout[i] : 0.f;
}

__global__ void build_fin_kernel(const int* __restrict__ sent, const int* __restrict__ bushou,
                                 const float* __restrict__ pinyin,
                                 const float* __restrict__ transw,
                                 const float* __restrict__ weizhi,
                                 const float* __restrict__ emb1, const float* __restrict__ emb2,
                                 _Float16* __restrict__ fin16) {
  int bl = blockIdx.x;     // B*L
  int d = threadIdx.x;     // 256
  float v;
  if (d < 100)       v = emb1[(size_t)sent[bl] * 100 + d];
  else if (d < 200)  v = emb2[(size_t)bushou[bl] * 100 + (d - 100)];
  else if (d < 220)  v = pinyin[(size_t)bl * 20 + (d - 200)];
  else if (d < 238)  v = transw[(size_t)bl * 18 + (d - 220)];
  else               v = weizhi[(size_t)bl * 18 + (d - 238)];
  fin16[(size_t)bl * DM_ + d] = (_Float16)v;
}

__global__ void zero_out_kernel(float* o) {
  if (threadIdx.x == 0 && blockIdx.x == 0) o[0] = 0.f;
}

// ---------------------------------------------------------------------------
extern "C" void kernel_launch(void* const* d_in, const int* in_sizes, int n_in,
                              void* d_out, int out_size, void* d_ws, size_t ws_size,
                              hipStream_t stream) {
  (void)in_sizes; (void)n_in; (void)out_size; (void)ws_size;
  const int*   sentences = (const int*)d_in[0];
  const int*   bushou    = (const int*)d_in[1];
  const float* pinyin    = (const float*)d_in[2];
  const float* weizhi    = (const float*)d_in[3];
  const float* transw    = (const float*)d_in[4];
  const int*   tags      = (const int*)d_in[5];
  const int*   lengths   = (const int*)d_in[6];
  const float* emb1      = (const float*)d_in[7];
  const float* emb2      = (const float*)d_in[8];
  const float* Wq = (const float*)d_in[9];  const float* bq = (const float*)d_in[10];
  const float* Wk = (const float*)d_in[11]; const float* bk = (const float*)d_in[12];
  const float* Wv = (const float*)d_in[13]; const float* bv = (const float*)d_in[14];
  const float* Wo = (const float*)d_in[15]; const float* bo = (const float*)d_in[16];
  const float* Wih_f = (const float*)d_in[17]; const float* Whh_f = (const float*)d_in[18];
  const float* bih_f = (const float*)d_in[19]; const float* bhh_f = (const float*)d_in[20];
  const float* Wih_b = (const float*)d_in[21]; const float* Whh_b = (const float*)d_in[22];
  const float* bih_b = (const float*)d_in[23]; const float* bhh_b = (const float*)d_in[24];
  const float* Wout = (const float*)d_in[25]; const float* bout = (const float*)d_in[26];
  const float* transitions = (const float*)d_in[27];
  const float* h0 = (const float*)d_in[28];  const float* c0 = (const float*)d_in[29];
  float* out = (float*)d_out;

  // ---- workspace bump allocator ----
  char* base = (char*)d_ws;
  size_t off = 0;
  auto alloc = [&](size_t bytes) -> void* {
    void* p = base + off;
    off = (off + bytes + 255) & ~(size_t)255;
    return p;
  };
  const int ML = B_ * L_;                       // 16384
  const size_t SZ_ACT = (size_t)ML * DM_ * 2;   // f16 activation (8 MB)

  _Float16* fin16  = (_Float16*)alloc(SZ_ACT);
  _Float16* qh     = (_Float16*)alloc(SZ_ACT);
  _Float16* kh     = (_Float16*)alloc(SZ_ACT);
  _Float16* vh     = (_Float16*)alloc(SZ_ACT);
  float*    scores = (float*)alloc((size_t)B_ * H_ * L_ * L_ * 4);     // 64 MB
  _Float16* probs  = (_Float16*)alloc((size_t)B_ * H_ * L_ * L_ * 2);  // 32 MB
  _Float16* xh     = (_Float16*)alloc(SZ_ACT);
  _Float16* xoh    = (_Float16*)alloc(SZ_ACT);
  _Float16* Wq16   = (_Float16*)alloc((size_t)DM_ * DM_ * 2);
  _Float16* Wk16   = (_Float16*)alloc((size_t)DM_ * DM_ * 2);
  _Float16* Wv16   = (_Float16*)alloc((size_t)DM_ * DM_ * 2);
  _Float16* Wo16   = (_Float16*)alloc((size_t)DM_ * DM_ * 2);
  _Float16* Wihf16 = (_Float16*)alloc((size_t)512 * DM_ * 2);
  _Float16* Wihb16 = (_Float16*)alloc((size_t)512 * DM_ * 2);
  _Float16* Whhf16 = (_Float16*)alloc((size_t)512 * HID2_ * 2);
  _Float16* Whhb16 = (_Float16*)alloc((size_t)512 * HID2_ * 2);
  _Float16* WoutP  = (_Float16*)alloc((size_t)NPAD_ * DM_ * 2);
  float*    boutP  = (float*)alloc(NPAD_ * 4);
  float*    bsum_f = (float*)alloc(512 * 4);
  float*    bsum_b = (float*)alloc(512 * 4);
  float*    grec   = (float*)alloc((size_t)2 * B_ * 512 * 4);
  // aliases over dead regions:
  float*    gi_f      = scores;            // scores dead after softmax
  float*    gi_b      = (float*)probs;     // probs dead after PV (gi computed after)
  _Float16* lstm_out  = fin16;             // fin dead after QKV GEMMs
  float*    logitsPad = (float*)qh;        // q dead after scores

  // ---- weight conversions / bias prep ----
  auto launch_f2h = [&](const float* s, _Float16* d, int n) {
    f2h_kernel<<<(n + 255) / 256, 256, 0, stream>>>(s, d, n);
  };
  launch_f2h(Wq, Wq16, DM_ * DM_);
  launch_f2h(Wk, Wk16, DM_ * DM_);
  launch_f2h(Wv, Wv16, DM_ * DM_);
  launch_f2h(Wo, Wo16, DM_ * DM_);
  launch_f2h(Wih_f, Wihf16, 512 * DM_);
  launch_f2h(Wih_b, Wihb16, 512 * DM_);
  launch_f2h(Whh_f, Whhf16, 512 * HID2_);
  launch_f2h(Whh_b, Whhb16, 512 * HID2_);
  pad_wout_kernel<<<(NPAD_ * DM_ + 255) / 256, 256, 0, stream>>>(Wout, bout, WoutP, boutP);
  bias_sum_kernel<<<2, 256, 0, stream>>>(bih_f, bhh_f, bsum_f, 512);
  bias_sum_kernel<<<2, 256, 0, stream>>>(bih_b, bhh_b, bsum_b, 512);

  // ---- feature concat ----
  build_fin_kernel<<<ML, 256, 0, stream>>>(sentences, bushou, pinyin, transw, weizhi,
                                           emb1, emb2, fin16);

  // ---- attention ----
  auto launch_gemm = [&](const _Float16* A, const _Float16* W, const float* bias,
                         float* Cf, _Float16* Ch, int M, int N, int K) {
    int blocks = ((M >> 4) * (N >> 4) + 7) / 8;
    gemm_xw_kernel<<<blocks, 256, 0, stream>>>(A, W, bias, Cf, Ch, M, N, K);
  };
  launch_gemm(fin16, Wq16, bq, nullptr, qh, ML, DM_, DM_);
  launch_gemm(fin16, Wk16, bk, nullptr, kh, ML, DM_, DM_);
  launch_gemm(fin16, Wv16, bv, nullptr, vh, ML, DM_, DM_);

  attn_scores_kernel<<<dim3(32, B_ * H_), 256, 0, stream>>>(qh, kh, scores);
  softmax_rows_kernel<<<B_ * H_ * L_, 256, 0, stream>>>(scores, probs);
  attn_pv_kernel<<<dim3(8, B_ * H_), 256, 0, stream>>>(probs, vh, xh);
  launch_gemm(xh, Wo16, bo, nullptr, xoh, ML, DM_, DM_);

  // ---- LSTM input projections (batch-parallel) ----
  launch_gemm(xoh, Wihf16, bsum_f, gi_f, nullptr, ML, 512, DM_);
  launch_gemm(xoh, Wihb16, bsum_b, gi_b, nullptr, ML, 512, DM_);

  // ---- sequential recurrence: 1 persistent block per direction,
  //      Whh async-staged into 128 KB dynamic LDS ----
  lstm_recurrent_kernel<<<2, 512, 512 * HID2_ * 2, stream>>>(
      gi_f, gi_b, Whhf16, Whhb16, h0, c0, grec, lstm_out);

  // ---- logits (padded N=32) ----
  launch_gemm(lstm_out, WoutP, boutP, logitsPad, nullptr, ML, NPAD_, DM_);

  // ---- CRF ----
  zero_out_kernel<<<1, 32, 0, stream>>>(out);
  crf_kernel<<<B_, 32, 0, stream>>>(logitsPad, tags, lengths, transitions, out);
}
